// BatterySOHForecaster_4183298146498
// MI455X (gfx1250) — matrix operand order repaired and checked by reading, not compiled, not run on hardware
//
#include <hip/hip_runtime.h>

// ---------------------------------------------------------------------------
// CDNA5 (gfx1250) WMMA building blocks
// ---------------------------------------------------------------------------
typedef __attribute__((ext_vector_type(16))) _Float16 v16h;
typedef __attribute__((ext_vector_type(8)))  float    v8f;

static __device__ __forceinline__ v8f wmma32(v16h a, v16h b, v8f c) {
  return __builtin_amdgcn_wmma_f32_16x16x32_f16(false, a, false, b, (short)0, c,
                                                false, false);
}

// A fragment: 16x32 f16, row-major source with leading dim ld.
// Layout used here (contiguous k = 16h + s): lanes 0-15 -> M=0..15 (K 0..15), lanes 16-31 -> (K 16..31)
static __device__ __forceinline__ v16h load_a32(const _Float16* p, int ld) {
  int lane = threadIdx.x & 31;
  int m = lane & 15, hf = lane >> 4;
  v16h a;
#pragma unroll
  for (int v = 0; v < 8; ++v) {
    int k0 = hf * 16 + 2 * v;
    a[2 * v]     = p[m * ld + k0];
    a[2 * v + 1] = p[m * ld + k0 + 1];
  }
  return a;
}

// B fragment: 32x16 f16, row-major (K x N) source with leading dim ld.
// lane: n = lane&15, K-half = lane>>4; VGPR v holds K = kh*16+2v, +1
static __device__ __forceinline__ v16h load_b32(const _Float16* p, int ld) {
  int lane = threadIdx.x & 31;
  int n = lane & 15, kh = lane >> 4;
  v16h b;
#pragma unroll
  for (int v = 0; v < 8; ++v) {
    b[2 * v]     = p[(kh * 16 + 2 * v) * ld + n];
    b[2 * v + 1] = p[(kh * 16 + 2 * v + 1) * ld + n];
  }
  return b;
}

// C/D: lane n = lane&15, rows rb..rb+7 with rb = (lane>>4)*8
static __device__ __forceinline__ void store_c16(float* p, int ld, v8f c) {
  int lane = threadIdx.x & 31;
  int n = lane & 15, rb = (lane >> 4) * 8;
#pragma unroll
  for (int v = 0; v < 8; ++v) p[(rb + v) * ld + n] = c[v];
}

static __device__ __forceinline__ float sigmoidf_(float x) {
  return 1.0f / (1.0f + __expf(-x));
}

// ---------------------------------------------------------------------------
// Weight prep kernels
// ---------------------------------------------------------------------------
// dst (Kp x Np) f16 = transpose(src (N x K)) zero-padded
__global__ void transpose_pad_f16(const float* src, _Float16* dst, int N, int K,
                                  int Kp, int Np) {
  int i = blockIdx.x * 256 + threadIdx.x;
  if (i >= Kp * Np) return;
  int k = i / Np, n = i % Np;
  dst[i] = (k < K && n < N) ? (_Float16)src[(size_t)n * K + k] : (_Float16)0.f;
}

// dst (M x Kp) f16 = src (M x K) zero-padded along K (A-matrix style)
__global__ void pad_f16(const float* src, _Float16* dst, int M, int K, int Kp) {
  int i = blockIdx.x * 256 + threadIdx.x;
  if (i >= M * Kp) return;
  int m = i / Kp, k = i % Kp;
  dst[i] = (k < K) ? (_Float16)src[(size_t)m * K + k] : (_Float16)0.f;
}

__global__ void pad_bias(const float* src, float* dst, int R, int n, int np) {
  int i = blockIdx.x * 256 + threadIdx.x;
  if (i >= R * np) return;
  int r = i / np, j = i % np;
  dst[i] = (j < n) ? src[(size_t)r * n + j] : 0.f;
}

// Xcat f16 (12800 x 32): [cycle_stats(16), soh(1), pad]
__global__ void build_xcat(const float* cyc, const float* soh, _Float16* X) {
  int i = blockIdx.x * 256 + threadIdx.x;
  if (i >= 12800 * 32) return;
  int row = i >> 5, k = i & 31;
  float v = 0.f;
  if (k < 16) v = cyc[(size_t)row * 16 + k];
  else if (k == 16) v = soh[row];
  X[i] = (_Float16)v;
}

// Xf f16 (3072 x 32): future_ops * mask, pad
__global__ void build_xf(const float* fo, const float* fm, _Float16* X) {
  int i = blockIdx.x * 256 + threadIdx.x;
  if (i >= 3072 * 32) return;
  int row = i >> 5, k = i & 31;
  float v = (k < 4) ? fo[(size_t)row * 4 + k] * fm[row] : 0.f;
  X[i] = (_Float16)v;
}

__global__ void cycle_mask_kernel(const float* qv_masks, float* cmask) {
  int i = blockIdx.x * 256 + threadIdx.x;
  if (i >= 12800) return;
  float s = 0.f;
  for (int w = 0; w < 100; ++w) s += qv_masks[(size_t)i * 100 + w];
  cmask[i] = (s > 0.f) ? 1.f : 0.f;
}

// ---------------------------------------------------------------------------
// Generic WMMA GEMM: C[M,N] = A[M,K](f16) x B[K,N](f16) (+bias, relu, f16 copy)
// grid (M/16, N/16, E); block 32 (one wave). K multiple of 32.
// ---------------------------------------------------------------------------
__global__ void wmma_gemm_kernel(const _Float16* Am, const _Float16* Bm,
                                 const float* biasp, float* Cp, _Float16* Chp,
                                 int M, int N, int K, long long sAe,
                                 long long sBe, long long sbe, long long sCe,
                                 int relu) {
  int mt = blockIdx.x, nt = blockIdx.y, e = blockIdx.z;
  Am += (size_t)e * sAe;
  Bm += (size_t)e * sBe;
  Cp += (size_t)e * sCe;
  if (Chp) Chp += (size_t)e * sCe;
  if (biasp) biasp += (size_t)e * sbe;

  v8f acc = {};
  for (int kc = 0; kc < K; kc += 32) {
    if (kc + 32 < K)
      __builtin_prefetch(Bm + (size_t)(kc + 32) * N + nt * 16, 0, 3);
    v16h a = load_a32(Am + (size_t)mt * 16 * K + kc, K);
    v16h b = load_b32(Bm + (size_t)kc * N + nt * 16, N);
    acc = wmma32(a, b, acc);
  }
  int lane = threadIdx.x & 31;
  int n = lane & 15, rb = (lane >> 4) * 8;
#pragma unroll
  for (int v = 0; v < 8; ++v) {
    float x = acc[v] + (biasp ? biasp[nt * 16 + n] : 0.f);
    if (relu) x = fmaxf(x, 0.f);
    size_t idx = (size_t)(mt * 16 + rb + v) * N + nt * 16 + n;
    Cp[idx] = x;
    if (Chp) Chp[idx] = (_Float16)x;
  }
}

// ---------------------------------------------------------------------------
// GRU kernel. One workgroup per 16-row batch tile; 8 waves; hidden 256,
// gates 768. gi computed in-kernel from X (rows x 32 f16).
// Dynamic LDS: gAll f32[16*768] | giN f32[16*256] | hF f32[16*256]
//              | hA f16[16*256] | xA f16[16*32]
// ---------------------------------------------------------------------------
__global__ void gru_kernel(const _Float16* X, const _Float16* Bih,
                           const _Float16* BhhT, const float* bih,
                           const float* bhh, float* h_out, int S) {
  extern __shared__ char smem[];
  float* gAll = (float*)smem;                       // 16*768
  float* giN = gAll + 16 * 768;                     // 16*256
  float* hF = giN + 16 * 256;                       // 16*256
  _Float16* hA = (_Float16*)(hF + 16 * 256);        // 16*256
  _Float16* xA = hA + 16 * 256;                     // 16*32

  const int b0 = blockIdx.x * 16;
  const int wave = threadIdx.x >> 5;

  for (int i = threadIdx.x; i < 16 * 256; i += 256) {
    hF[i] = 0.f;
    hA[i] = (_Float16)0.f;
  }
  __syncthreads();

  for (int t = 0; t < S; ++t) {
    // stage x_t tile (16 x 32) into LDS
    for (int i = threadIdx.x; i < 16 * 32; i += 256) {
      int m = i >> 5, k = i & 31;
      xA[i] = X[((size_t)(b0 + m) * S + t) * 32 + k];
    }
    __syncthreads();

    // GEMMs: 48 gate-column tiles (768 = 48*16). nt<32 => r,z gates (ih+hh
    // combined); nt>=32 => n gate (keep ih and hh separate).
    for (int nt = wave; nt < 48; nt += 8) {
      if (nt < 32) {
        v8f acc = {};
        acc = wmma32(load_a32(xA, 32), load_b32(Bih + nt * 16, 768), acc);
        for (int kc = 0; kc < 8; ++kc)
          acc = wmma32(load_a32(hA + kc * 32, 256),
                       load_b32(BhhT + (size_t)kc * 32 * 768 + nt * 16, 768),
                       acc);
        store_c16(gAll + nt * 16, 768, acc);
      } else {
        v8f agi = {}, agh = {};
        agi = wmma32(load_a32(xA, 32), load_b32(Bih + nt * 16, 768), agi);
        for (int kc = 0; kc < 8; ++kc)
          agh = wmma32(load_a32(hA + kc * 32, 256),
                       load_b32(BhhT + (size_t)kc * 32 * 768 + nt * 16, 768),
                       agh);
        store_c16(giN + (nt - 32) * 16, 256, agi);
        store_c16(gAll + nt * 16, 768, agh);
      }
    }
    __syncthreads();

    // elementwise gate math: thread j handles column j, rows m=0..15
    int j = threadIdx.x;
    float bihr = bih[j] + bhh[j];
    float bihz = bih[256 + j] + bhh[256 + j];
    float bin = bih[512 + j];
    float bhn = bhh[512 + j];
    for (int m = 0; m < 16; ++m) {
      float r = sigmoidf_(gAll[m * 768 + j] + bihr);
      float z = sigmoidf_(gAll[m * 768 + 256 + j] + bihz);
      float ghn = gAll[m * 768 + 512 + j] + bhn;
      float gin = giN[m * 256 + j] + bin;
      float nn = tanhf(gin + r * ghn);
      float h = (1.f - z) * nn + z * hF[m * 256 + j];
      hF[m * 256 + j] = h;
      hA[m * 256 + j] = (_Float16)h;
    }
    __syncthreads();
  }

  for (int i = threadIdx.x; i < 16 * 256; i += 256)
    h_out[(size_t)(b0 + (i >> 8)) * 256 + (i & 255)] = hF[i];
}

// ---------------------------------------------------------------------------
// qv encoder: per (b,s) sample; conv1 (1->8,3x5) VALU; conv2 (8->16,3x5) as
// im2col WMMA GEMM (M=16 ch, K=120->128, N=800 positions); fused relu+pool.
// block 256 (8 waves).
// ---------------------------------------------------------------------------
__global__ void qv_encoder_kernel(const float* qv_maps, const float* k1,
                                  const float* b1, const _Float16* W2g,
                                  const float* b2, _Float16* out) {
  __shared__ float xs[800];
  __shared__ _Float16 h1[8 * 800];
  __shared__ float pooled[64];
  int bs = blockIdx.x;
  const float* xp = qv_maps + (size_t)bs * 800;
  for (int i = threadIdx.x; i < 800; i += 256) xs[i] = xp[i];
  if (threadIdx.x < 64) pooled[threadIdx.x] = 0.f;
  __syncthreads();

  // conv1 + relu -> h1 f16
  for (int idx = threadIdx.x; idx < 6400; idx += 256) {
    int oc = idx / 800, rem = idx % 800, r = rem / 100, c = rem % 100;
    float acc = b1[oc];
    for (int kr = 0; kr < 3; ++kr) {
      int rr = r - 1 + kr;
      if (rr < 0 || rr >= 8) continue;
      for (int kc = 0; kc < 5; ++kc) {
        int cc = c - 2 + kc;
        if (cc < 0 || cc >= 100) continue;
        acc += k1[oc * 15 + kr * 5 + kc] * xs[rr * 100 + cc];
      }
    }
    h1[idx] = (_Float16)fmaxf(acc, 0.f);
  }
  __syncthreads();

  int lane = threadIdx.x & 31, wave = threadIdx.x >> 5;
  int n = lane & 15, kh = lane >> 4, rb = kh * 8;
  for (int tile = wave; tile < 50; tile += 8) {
    int p = tile * 16 + n, r = p / 100, c = p % 100;
    v8f acc = {};
    for (int chunk = 0; chunk < 4; ++chunk) {
      v16h a = load_a32(W2g + chunk * 32, 128);
      v16h bf;
#pragma unroll
      for (int v = 0; v < 8; ++v) {
#pragma unroll
        for (int u = 0; u < 2; ++u) {
          int k = chunk * 32 + kh * 16 + 2 * v + u;
          float val = 0.f;
          if (k < 120) {
            int ci = k / 15, kr = (k % 15) / 5, kc = k % 5;
            int rr = r - 1 + kr, cc = c - 2 + kc;
            if (rr >= 0 && rr < 8 && cc >= 0 && cc < 100)
              val = (float)h1[ci * 800 + rr * 100 + cc];
          }
          bf[2 * v + u] = (_Float16)val;
        }
      }
      acc = wmma32(a, bf, acc);
    }
    int g = c / 25;
#pragma unroll
    for (int v = 0; v < 8; ++v) {
      int ch = rb + v;
      float val = fmaxf(acc[v] + b2[ch], 0.f);
      atomicAdd(&pooled[ch * 4 + g], val);
    }
  }
  __syncthreads();
  if (threadIdx.x < 64)
    out[(size_t)bs * 64 + threadIdx.x] =
        (_Float16)(pooled[threadIdx.x] / 200.f);
}

// ---------------------------------------------------------------------------
// 1D curve encoder (pc L=48, rx L=32): conv1 VALU, conv2 (8->16,k5) WMMA
// (M=16, K=40->64, N=L). block 32 (one wave).
// ---------------------------------------------------------------------------
__global__ void curve_encoder_kernel(const float* vals, const float* k1,
                                     const float* b1, const _Float16* W2g,
                                     const float* b2, _Float16* out, int L) {
  __shared__ float xs[48];
  __shared__ _Float16 h1[8 * 48];
  __shared__ float pooled[64];
  int bs = blockIdx.x;
  int lane = threadIdx.x;
  const float* xp = vals + (size_t)bs * L;
  for (int i = lane; i < L; i += 32) xs[i] = xp[i];
  for (int i = lane; i < 64; i += 32) pooled[i] = 0.f;
  __syncthreads();

  for (int idx = lane; idx < 8 * L; idx += 32) {
    int oc = idx / L, c = idx % L;
    float acc = b1[oc];
    for (int kc = 0; kc < 5; ++kc) {
      int cc = c - 2 + kc;
      if (cc >= 0 && cc < L) acc += k1[oc * 5 + kc] * xs[cc];
    }
    h1[idx] = (_Float16)fmaxf(acc, 0.f);
  }
  __syncthreads();

  int n = lane & 15, kh = lane >> 4, rb = kh * 8;
  int Gsz = L / 4, ntiles = L / 16;
  for (int tile = 0; tile < ntiles; ++tile) {
    int c = tile * 16 + n;
    v8f acc = {};
    for (int chunk = 0; chunk < 2; ++chunk) {
      v16h a = load_a32(W2g + chunk * 32, 64);
      v16h bf;
#pragma unroll
      for (int v = 0; v < 8; ++v) {
#pragma unroll
        for (int u = 0; u < 2; ++u) {
          int k = chunk * 32 + kh * 16 + 2 * v + u;
          float val = 0.f;
          if (k < 40) {
            int ci = k / 5, kc = k % 5, cc = c - 2 + kc;
            if (cc >= 0 && cc < L) val = (float)h1[ci * L + cc];
          }
          bf[2 * v + u] = (_Float16)val;
        }
      }
      acc = wmma32(a, bf, acc);
    }
#pragma unroll
    for (int v = 0; v < 8; ++v) {
      int ch = rb + v;
      float val = fmaxf(acc[v] + b2[ch], 0.f);
      atomicAdd(&pooled[ch * 4 + c / Gsz], val);
    }
  }
  __syncthreads();
  for (int i = lane; i < 64; i += 32)
    out[(size_t)bs * 64 + i] = (_Float16)(pooled[i] / (float)Gsz);
}

// masked mean over S of (12800 x 64) f16 features -> (128 x 64) f16
__global__ void masked_hbar_kernel(const _Float16* feats, const float* mask,
                                   _Float16* hbar, int S) {
  int b = blockIdx.x, j = threadIdx.x;  // block 64
  float acc = 0.f, wsum = 0.f;
  for (int s = 0; s < S; ++s) {
    float m = mask[(size_t)b * S + s];
    acc += m * (float)feats[((size_t)b * S + s) * 64 + j];
    wsum += m;
  }
  hbar[(size_t)b * 64 + j] = (_Float16)(acc / fmaxf(wsum, 1e-6f));
}

// ---------------------------------------------------------------------------
// Small per-batch feature assembly: meta, state, naive, rin(83), fin(25)
// ---------------------------------------------------------------------------
__global__ void small_feats_kernel(
    const float* soh_seq, const float* anchor_soh, const float* anchor_phys,
    const float* phys, const float* phys_mask, const float* op_seq,
    const float* fut_ops, const float* pc, const float* pc_mask,
    const float* rx, const float* rx_mask, const int* meta_ids,
    const float* meta_emb, float* meta, float* state, float* naive, float* rin,
    float* fin) {
  const int S = 100;
  int b = blockIdx.x, lane = threadIdx.x;  // block 32
  __shared__ float st[3];
  __shared__ float mloc[16];
  __shared__ float rloc[96];
  for (int i = lane; i < 96; i += 32) rloc[i] = 0.f;
  if (lane < 16) {
    float m = 0.f;
    for (int i = 0; i < 4; ++i) {
      int id = meta_ids[b * 4 + i];
      id = id < 0 ? 0 : (id > 511 ? 511 : id);
      m += meta_emb[id * 16 + lane];
    }
    mloc[lane] = m * 0.25f;
  }
  if (lane == 0) {
    float s99 = soh_seq[(size_t)b * S + 99];
    float s98 = soh_seq[(size_t)b * S + 98];
    float s97 = soh_seq[(size_t)b * S + 97];
    st[0] = anchor_soh[b];
    st[1] = s99 - s98;
    st[2] = s99 - 2.f * s98 + s97;
  }
  __syncthreads();

  if (lane < 8) {  // physics stats (curve block)
    float mu = 0.f;
    for (int s = 0; s < S; ++s) mu += phys[((size_t)b * S + s) * 8 + lane];
    mu /= (float)S;
    float m2 = 0.f;
    for (int s = 0; s < S; ++s) {
      float d = phys[((size_t)b * S + s) * 8 + lane] - mu;
      m2 += d * d;
    }
    rloc[3 + lane] = anchor_phys[b * 8 + lane];
    rloc[11 + lane] = mu;
    rloc[19 + lane] = sqrtf(m2 / (float)(S - 1));
    rloc[27 + lane] = phys[((size_t)b * S + 99) * 8 + lane] -
                      phys[((size_t)b * S + 0) * 8 + lane];
  }
  if (lane == 8) {  // partial_g
    const float* a = pc + ((size_t)b * S + 99) * 48;
    float mu = 0.f;
    for (int i = 0; i < 48; ++i) mu += a[i];
    mu /= 48.f;
    float m2 = 0.f;
    for (int i = 0; i < 48; ++i) {
      float d = a[i] - mu;
      m2 += d * d;
    }
    float mm = 0.f;
    for (int s = 0; s < S; ++s) mm += pc_mask[(size_t)b * S + s];
    rloc[35] = mu;
    rloc[36] = sqrtf(m2 / 47.f);
    rloc[37] = a[47];
    rloc[38] = mm / (float)S;
  }
  if (lane == 9) {  // relax_g
    const float* a = rx + ((size_t)b * S + 99) * 32;
    float mu = 0.f;
    for (int i = 0; i < 32; ++i) mu += a[i];
    mu /= 32.f;
    float m2 = 0.f;
    for (int i = 0; i < 32; ++i) {
      float d = a[i] - mu;
      m2 += d * d;
    }
    float mm = 0.f;
    for (int s = 0; s < S; ++s) mm += rx_mask[(size_t)b * S + s];
    rloc[39] = mu;
    rloc[40] = sqrtf(m2 / 31.f);
    rloc[41] = a[31] - a[0];
    rloc[42] = mm / (float)S;
  }
  if (lane >= 10 && lane < 16) {  // op_seq stats
    int d = lane - 10;
    float mu = 0.f;
    for (int s = 0; s < S; ++s) mu += op_seq[((size_t)b * S + s) * 6 + d];
    mu /= (float)S;
    float m2 = 0.f;
    for (int s = 0; s < S; ++s) {
      float dv = op_seq[((size_t)b * S + s) * 6 + d] - mu;
      m2 += dv * dv;
    }
    rloc[43 + d] = mu;
    rloc[49 + d] = sqrtf(m2 / (float)(S - 1));
  }
  if (lane >= 16 && lane < 20) {  // future_ops mean
    int d = lane - 16;
    float mu = 0.f;
    for (int t = 0; t < 24; ++t) mu += fut_ops[((size_t)b * 24 + t) * 4 + d];
    rloc[55 + d] = mu / 24.f;
  }
  if (lane == 20) {  // avail
    float pm = 0.f, cm = 0.f, rm = 0.f;
    for (int s = 0; s < S; ++s) {
      pm += phys_mask[(size_t)b * S + s];
      cm += pc_mask[(size_t)b * S + s];
      rm += rx_mask[(size_t)b * S + s];
    }
    rloc[80] = pm / (float)S;
    rloc[81] = cm / (float)S;
    rloc[82] = rm / (float)S;
  }
  if (lane == 21) {
    rloc[0] = st[0];
    rloc[1] = st[1];
    rloc[2] = st[2];
  }
  if (lane < 16) rloc[59 + lane] = mloc[lane];  // meta slot (mloc set pre-sync)
  __syncthreads();

  for (int i = lane; i < 83; i += 32) rin[(size_t)b * 83 + i] = rloc[i];
  if (lane < 16) meta[b * 16 + lane] = mloc[lane];
  if (lane < 3) state[b * 3 + lane] = st[lane];
  if (lane < 24) naive[b * 24 + lane] = st[0] + st[1] * (float)(lane + 1);
  for (int i = lane; i < 25; i += 32) {
    float v = 0.f;
    if (i >= 4 && i < 7) v = st[i - 4];
    else if (i == 7) v = rloc[80];
    else if (i == 8) v = rloc[81];
    else if (i >= 9) v = mloc[i - 9];
    fin[(size_t)b * 25 + i] = v;
  }
}

// gen MLP input assembly: f16 (128 x 320) = [h_cycle(256), meta(16), tsfm(32), pad]
__global__ void gen_in_assembly(const float* hc, const float* meta,
                                const float* tsfm, _Float16* A) {
  int i = blockIdx.x * 256 + threadIdx.x;
  if (i >= 128 * 320) return;
  int b = i / 320, j = i % 320;
  float v = 0.f;
  if (j < 256) v = hc[b * 256 + j];
  else if (j < 272) v = meta[b * 16 + (j - 256)];
  else if (j < 304) v = tsfm[b * 32 + (j - 272)];
  A[i] = (_Float16)v;
}

// expert input assembly: f16 (128 x 1408)
__global__ void expert_in_assembly(const float* hc, const float* hqv,
                                   const float* hpc, const float* hrx,
                                   const float* hfut, const float* meta,
                                   const float* tsfm, const float* state,
                                   const float* gen_pred, const float* naive,
                                   _Float16* A) {
  int i = blockIdx.x * 256 + threadIdx.x;
  if (i >= 128 * 1408) return;
  int b = i / 1408, j = i % 1408;
  float v = 0.f;
  if (j < 256) v = hc[b * 256 + j];
  else if (j < 512) v = hqv[b * 256 + j - 256];
  else if (j < 768) v = hpc[b * 256 + j - 512];
  else if (j < 1024) v = hrx[b * 256 + j - 768];
  else if (j < 1280) v = hfut[b * 256 + j - 1024];
  else if (j < 1296) v = meta[b * 16 + j - 1280];
  else if (j < 1328) v = tsfm[b * 32 + j - 1296];
  else if (j < 1331) v = state[b * 3 + j - 1328];
  else if (j < 1355) v = gen_pred[b * 32 + (j - 1331)];
  else if (j < 1379) v = naive[b * 24 + (j - 1355)];
  else if (j < 1403) v = gen_pred[b * 32 + (j - 1379)] - naive[b * 24 + (j - 1379)];
  A[i] = (_Float16)v;
}

// router softmax + top-2 gating + fusion
__global__ void final_kernel(const float* rin, const float* fin,
                             const float* rt_W, const float* rt_b,
                             const float* fu_W, const float* fu_b,
                             const float* expert_outs, const float* gen_pred,
                             const float* naive, float* out) {
  int b = blockIdx.x, lane = threadIdx.x;  // block 32
  float probs[8];
  float mx = -1e30f;
  for (int e = 0; e < 8; ++e) {
    float acc = rt_b[e];
    for (int i = 0; i < 83; ++i) acc += rt_W[e * 83 + i] * rin[(size_t)b * 83 + i];
    probs[e] = acc;
    mx = fmaxf(mx, acc);
  }
  float ssum = 0.f;
  for (int e = 0; e < 8; ++e) {
    probs[e] = __expf(probs[e] - mx);
    ssum += probs[e];
  }
  for (int e = 0; e < 8; ++e) probs[e] /= ssum;
  int i0 = 0;
  float v0 = probs[0];
  for (int e = 1; e < 8; ++e)
    if (probs[e] > v0) { v0 = probs[e]; i0 = e; }
  int i1 = (i0 == 0) ? 1 : 0;
  float v1 = probs[i1];
  for (int e = 0; e < 8; ++e)
    if (e != i0 && probs[e] > v1) { v1 = probs[e]; i1 = e; }
  float gs = v0 + v1, g0 = v0 / gs, g1 = v1 / gs;

  float fl[4];
  float fmx = -1e30f;
  for (int o = 0; o < 4; ++o) {
    float acc = fu_b[o];
    for (int i = 0; i < 25; ++i) acc += fu_W[o * 25 + i] * fin[(size_t)b * 25 + i];
    fl[o] = acc;
    fmx = fmaxf(fmx, acc);
  }
  float fs = 0.f;
  for (int o = 0; o < 4; ++o) {
    fl[o] = __expf(fl[o] - fmx);
    fs += fl[o];
  }
  for (int o = 0; o < 4; ++o) fl[o] /= fs;

  if (lane < 24) {
    float ep = g0 * expert_outs[((size_t)i0 * 128 + b) * 32 + lane] +
               g1 * expert_outs[((size_t)i1 * 128 + b) * 32 + lane];
    float gp = gen_pred[(size_t)b * 32 + lane];
    float nv = naive[(size_t)b * 24 + lane];
    out[(size_t)b * 24 + lane] =
        fl[0] * gp + fl[1] * ep + fl[2] * nv + fl[3] * 0.5f * (gp + ep);
  }
}

// ---------------------------------------------------------------------------
// Host orchestration
// ---------------------------------------------------------------------------
extern "C" void kernel_launch(void* const* d_in, const int* in_sizes, int n_in,
                              void* d_out, int out_size, void* d_ws,
                              size_t ws_size, hipStream_t stream) {
  (void)in_sizes; (void)n_in; (void)out_size; (void)ws_size;
  char* ws = (char*)d_ws;
  size_t off = 0;
  auto alloc = [&](size_t bytes) -> void* {
    void* p = ws + off;
    off += (bytes + 255) & ~(size_t)255;
    return p;
  };

  // inputs
  const float* cycle_stats = (const float*)d_in[0];
  const float* soh_seq = (const float*)d_in[1];
  const float* anchor_soh = (const float*)d_in[2];
  const float* qv_maps = (const float*)d_in[3];
  const float* qv_masks = (const float*)d_in[4];
  const float* pc_curves = (const float*)d_in[5];
  const float* pc_mask = (const float*)d_in[6];
  const float* rx_curves = (const float*)d_in[7];
  const float* rx_mask = (const float*)d_in[8];
  const float* future_ops = (const float*)d_in[9];
  const float* future_ops_mask = (const float*)d_in[10];
  const float* anchor_phys = (const float*)d_in[11];
  const float* tsfm = (const float*)d_in[12];
  const float* phys = (const float*)d_in[13];
  const float* phys_mask = (const float*)d_in[14];
  const float* op_seq = (const float*)d_in[15];
  const int* meta_ids = (const int*)d_in[16];
  const float* cyc_Wih = (const float*)d_in[17];
  const float* cyc_Whh = (const float*)d_in[18];
  const float* cyc_bih = (const float*)d_in[19];
  const float* cyc_bhh = (const float*)d_in[20];
  const float* fut_Wih = (const float*)d_in[21];
  const float* fut_Whh = (const float*)d_in[22];
  const float* fut_bih = (const float*)d_in[23];
  const float* fut_bhh = (const float*)d_in[24];
  const float* qv_k1 = (const float*)d_in[25];
  const float* qv_b1 = (const float*)d_in[26];
  const float* qv_k2 = (const float*)d_in[27];
  const float* qv_b2 = (const float*)d_in[28];
  const float* qv_Wp = (const float*)d_in[29];
  const float* qv_bp = (const float*)d_in[30];
  const float* pc_k1 = (const float*)d_in[31];
  const float* pc_b1 = (const float*)d_in[32];
  const float* pc_k2 = (const float*)d_in[33];
  const float* pc_b2 = (const float*)d_in[34];
  const float* pc_Wp = (const float*)d_in[35];
  const float* pc_bp = (const float*)d_in[36];
  const float* rx_k1 = (const float*)d_in[37];
  const float* rx_b1 = (const float*)d_in[38];
  const float* rx_k2 = (const float*)d_in[39];
  const float* rx_b2 = (const float*)d_in[40];
  const float* rx_Wp = (const float*)d_in[41];
  const float* rx_bp = (const float*)d_in[42];
  const float* meta_emb = (const float*)d_in[43];
  const float* gen_W1 = (const float*)d_in[44];
  const float* gen_b1 = (const float*)d_in[45];
  const float* gen_W2 = (const float*)d_in[46];
  const float* gen_b2 = (const float*)d_in[47];
  const float* exp_W1 = (const float*)d_in[48];
  const float* exp_b1 = (const float*)d_in[49];
  const float* exp_W2 = (const float*)d_in[50];
  const float* exp_b2 = (const float*)d_in[51];
  const float* rt_W = (const float*)d_in[52];
  const float* rt_b = (const float*)d_in[53];
  const float* fu_W = (const float*)d_in[54];
  const float* fu_b = (const float*)d_in[55];

  // workspace (f16 buffers)
  _Float16* Bih_c = (_Float16*)alloc(32 * 768 * 2);
  _Float16* BhhT_c = (_Float16*)alloc(256 * 768 * 2);
  _Float16* Bih_f = (_Float16*)alloc(32 * 768 * 2);
  _Float16* BhhT_f = (_Float16*)alloc(256 * 768 * 2);
  _Float16* WpTqv = (_Float16*)alloc(64 * 256 * 2);
  _Float16* WpTpc = (_Float16*)alloc(64 * 256 * 2);
  _Float16* WpTrx = (_Float16*)alloc(64 * 256 * 2);
  _Float16* W2gqv = (_Float16*)alloc(16 * 128 * 2);
  _Float16* W2gpc = (_Float16*)alloc(16 * 64 * 2);
  _Float16* W2grx = (_Float16*)alloc(16 * 64 * 2);
  _Float16* genW1T = (_Float16*)alloc(320 * 256 * 2);
  _Float16* genW2T = (_Float16*)alloc(256 * 32 * 2);
  _Float16* expW1T = (_Float16*)alloc((size_t)8 * 1408 * 256 * 2);
  _Float16* expW2T = (_Float16*)alloc((size_t)8 * 256 * 32 * 2);
  _Float16* Xcat = (_Float16*)alloc((size_t)12800 * 32 * 2);
  _Float16* Xf = (_Float16*)alloc((size_t)3072 * 32 * 2);
  _Float16* qvh = (_Float16*)alloc((size_t)12800 * 64 * 2);
  _Float16* pch = (_Float16*)alloc((size_t)12800 * 64 * 2);
  _Float16* rxh = (_Float16*)alloc((size_t)12800 * 64 * 2);
  _Float16* hbarqv = (_Float16*)alloc(128 * 64 * 2);
  _Float16* hbarpc = (_Float16*)alloc(128 * 64 * 2);
  _Float16* hbarrx = (_Float16*)alloc(128 * 64 * 2);
  _Float16* geninA = (_Float16*)alloc(128 * 320 * 2);
  _Float16* g1h = (_Float16*)alloc(128 * 256 * 2);
  _Float16* expinA = (_Float16*)alloc(128 * 1408 * 2);
  _Float16* h1eh = (_Float16*)alloc((size_t)8 * 128 * 256 * 2);
  // f32 buffers
  float* cmask = (float*)alloc(12800 * 4);
  float* h_cycle = (float*)alloc(128 * 256 * 4);
  float* h_fut = (float*)alloc(128 * 256 * 4);
  float* hqv = (float*)alloc(128 * 256 * 4);
  float* hpc = (float*)alloc(128 * 256 * 4);
  float* hrx = (float*)alloc(128 * 256 * 4);
  float* meta = (float*)alloc(128 * 16 * 4);
  float* state = (float*)alloc(128 * 3 * 4);
  float* naive = (float*)alloc(128 * 24 * 4);
  float* rin = (float*)alloc(128 * 83 * 4);
  float* fin = (float*)alloc(128 * 25 * 4);
  float* g1 = (float*)alloc(128 * 256 * 4);
  float* gen_pred = (float*)alloc(128 * 32 * 4);
  float* h1e = (float*)alloc((size_t)8 * 128 * 256 * 4);
  float* expert_outs = (float*)alloc((size_t)8 * 128 * 32 * 4);
  float* genb2p = (float*)alloc(32 * 4);
  float* expb2p = (float*)alloc(8 * 32 * 4);

  auto tp = [&](const float* src, _Float16* dst, int N, int K, int Kp, int Np) {
    int total = Kp * Np;
    transpose_pad_f16<<<dim3((total + 255) / 256), dim3(256), 0, stream>>>(
        src, dst, N, K, Kp, Np);
  };

  // --- weight prep ---
  tp(cyc_Wih, Bih_c, 768, 17, 32, 768);
  tp(cyc_Whh, BhhT_c, 768, 256, 256, 768);
  tp(fut_Wih, Bih_f, 768, 4, 32, 768);
  tp(fut_Whh, BhhT_f, 768, 256, 256, 768);
  tp(qv_Wp, WpTqv, 256, 64, 64, 256);
  tp(pc_Wp, WpTpc, 256, 64, 64, 256);
  tp(rx_Wp, WpTrx, 256, 64, 64, 256);
  tp(gen_W1, genW1T, 256, 304, 320, 256);
  tp(gen_W2, genW2T, 24, 256, 256, 32);
  for (int e = 0; e < 8; ++e) {
    tp(exp_W1 + (size_t)e * 256 * 1403, expW1T + (size_t)e * 1408 * 256, 256,
       1403, 1408, 256);
    tp(exp_W2 + (size_t)e * 24 * 256, expW2T + (size_t)e * 256 * 32, 24, 256,
       256, 32);
  }
  pad_f16<<<dim3((16 * 128 + 255) / 256), dim3(256), 0, stream>>>(qv_k2, W2gqv, 16, 120, 128);
  pad_f16<<<dim3((16 * 64 + 255) / 256), dim3(256), 0, stream>>>(pc_k2, W2gpc, 16, 40, 64);
  pad_f16<<<dim3((16 * 64 + 255) / 256), dim3(256), 0, stream>>>(rx_k2, W2grx, 16, 40, 64);
  pad_bias<<<dim3(1), dim3(256), 0, stream>>>(gen_b2, genb2p, 1, 24, 32);
  pad_bias<<<dim3(1), dim3(256), 0, stream>>>(exp_b2, expb2p, 8, 24, 32);

  build_xcat<<<dim3(12800 * 32 / 256), dim3(256), 0, stream>>>(cycle_stats, soh_seq, Xcat);
  build_xf<<<dim3(3072 * 32 / 256), dim3(256), 0, stream>>>(future_ops, future_ops_mask, Xf);
  cycle_mask_kernel<<<dim3(50), dim3(256), 0, stream>>>(qv_masks, cmask);

  // --- GRUs (WMMA, sequential in S) ---
  const size_t gruLds = (size_t)(16 * 768 + 16 * 256 + 16 * 256) * 4 +
                        (size_t)(16 * 256 + 16 * 32) * 2;
  gru_kernel<<<dim3(8), dim3(256), gruLds, stream>>>(Xcat, Bih_c, BhhT_c,
                                                     cyc_bih, cyc_bhh, h_cycle, 100);
  gru_kernel<<<dim3(8), dim3(256), gruLds, stream>>>(Xf, Bih_f, BhhT_f,
                                                     fut_bih, fut_bhh, h_fut, 24);

  // --- encoders (conv2 as WMMA im2col GEMM) ---
  qv_encoder_kernel<<<dim3(12800), dim3(256), 0, stream>>>(qv_maps, qv_k1, qv_b1, W2gqv, qv_b2, qvh);
  curve_encoder_kernel<<<dim3(12800), dim3(32), 0, stream>>>(pc_curves, pc_k1, pc_b1, W2gpc, pc_b2, pch, 48);
  curve_encoder_kernel<<<dim3(12800), dim3(32), 0, stream>>>(rx_curves, rx_k1, rx_b1, W2grx, rx_b2, rxh, 32);

  // masked means (commuted before projection)
  masked_hbar_kernel<<<dim3(128), dim3(64), 0, stream>>>(qvh, cmask, hbarqv, 100);
  masked_hbar_kernel<<<dim3(128), dim3(64), 0, stream>>>(pch, pc_mask, hbarpc, 100);
  masked_hbar_kernel<<<dim3(128), dim3(64), 0, stream>>>(rxh, rx_mask, hbarrx, 100);

  // projections: (128x64) x (64x256) + bias
  wmma_gemm_kernel<<<dim3(8, 16, 1), dim3(32), 0, stream>>>(
      hbarqv, WpTqv, qv_bp, hqv, (_Float16*)nullptr, 128, 256, 64, 0, 0, 0, 0, 0);
  wmma_gemm_kernel<<<dim3(8, 16, 1), dim3(32), 0, stream>>>(
      hbarpc, WpTpc, pc_bp, hpc, (_Float16*)nullptr, 128, 256, 64, 0, 0, 0, 0, 0);
  wmma_gemm_kernel<<<dim3(8, 16, 1), dim3(32), 0, stream>>>(
      hbarrx, WpTrx, rx_bp, hrx, (_Float16*)nullptr, 128, 256, 64, 0, 0, 0, 0, 0);

  // small features
  small_feats_kernel<<<dim3(128), dim3(32), 0, stream>>>(
      soh_seq, anchor_soh, anchor_phys, phys, phys_mask, op_seq, future_ops,
      pc_curves, pc_mask, rx_curves, rx_mask, meta_ids, meta_emb, meta, state,
      naive, rin, fin);

  // gen MLP
  gen_in_assembly<<<dim3(128 * 320 / 256), dim3(256), 0, stream>>>(h_cycle, meta, tsfm, geninA);
  wmma_gemm_kernel<<<dim3(8, 16, 1), dim3(32), 0, stream>>>(
      geninA, genW1T, gen_b1, g1, g1h, 128, 256, 320, 0, 0, 0, 0, 1);
  wmma_gemm_kernel<<<dim3(8, 2, 1), dim3(32), 0, stream>>>(
      g1h, genW2T, genb2p, gen_pred, (_Float16*)nullptr, 128, 32, 256, 0, 0, 0, 0, 0);

  // expert MoE
  expert_in_assembly<<<dim3(128 * 1408 / 256), dim3(256), 0, stream>>>(
      h_cycle, hqv, hpc, hrx, h_fut, meta, tsfm, state, gen_pred, naive, expinA);
  wmma_gemm_kernel<<<dim3(8, 16, 8), dim3(32), 0, stream>>>(
      expinA, expW1T, exp_b1, h1e, h1eh, 128, 256, 1408, 0,
      (long long)1408 * 256, 256, (long long)128 * 256, 1);
  wmma_gemm_kernel<<<dim3(8, 2, 8), dim3(32), 0, stream>>>(
      h1eh, expW2T, expb2p, expert_outs, (_Float16*)nullptr, 128, 32, 256,
      (long long)128 * 256, (long long)256 * 32, 32, (long long)128 * 32, 0);

  // routing + fusion -> output (128 x 24) f32
  final_kernel<<<dim3(128), dim3(32), 0, stream>>>(
      rin, fin, rt_W, rt_b, fu_W, fu_b, expert_outs, gen_pred, naive,
      (float*)d_out);
}
